// Decode_40922448396939
// MI455X (gfx1250) — matrix operand order repaired and checked by reading, not compiled, not run on hardware
//
#include <hip/hip_runtime.h>
#include <hip/hip_bf16.h>

// ---------------- problem constants ----------------
#define E_EDGES   2048
#define DDIM      200
#define CCH       50
#define KDIM      10000        // C*D
#define KP        10016        // K padded to multiple of 32
#define NP        208          // 200 padded to multiple of 16 (13 tiles)
#define NNODES    40943
#define EPS_BN    1e-5f
#define EPS_COS   1e-8f

typedef _Float16 v8h  __attribute__((ext_vector_type(8)));
typedef _Float16 v16h __attribute__((ext_vector_type(16)));
typedef float    v8f  __attribute__((ext_vector_type(8)));

static __device__ inline v16h cat16(v8h lo, v8h hi) {
  return __builtin_shufflevector(lo, hi, 0,1,2,3,4,5,6,7,8,9,10,11,12,13,14,15);
}

static __device__ inline float wave_sum(float v) {
  #pragma unroll
  for (int off = 16; off > 0; off >>= 1) v += __shfl_xor(v, off, 32);
  return v;
}

// ---------------- small utility kernels ----------------
__global__ void zero_f32(float* __restrict__ p, int n) {
  int i = blockIdx.x * blockDim.x + threadIdx.x;
  if (i < n) p[i] = 0.0f;
}

__global__ void zero_f32x4(float4* __restrict__ p, int n4) {
  int i = blockIdx.x * blockDim.x + threadIdx.x;
  if (i < n4) p[i] = make_float4(0.f, 0.f, 0.f, 0.f);
}

// fc_w [200,10000] f32 -> B_f16 [NP][KP] f16 (zero padded)
__global__ void convert_b_kernel(const float* __restrict__ fcw,
                                 _Float16* __restrict__ B16) {
  long total = (long)NP * KP;
  long i = (long)blockIdx.x * blockDim.x + threadIdx.x;
  if (i >= total) return;
  int n = (int)(i / KP);
  int k = (int)(i - (long)n * KP);
  float v = (n < DDIM && k < KDIM) ? fcw[(long)n * KDIM + k] : 0.0f;
  B16[i] = (_Float16)v;
}

// ---------------- conv pass 1: per-channel sum/sumsq over (E,D) ----------------
template<int CIN>
__global__ void __launch_bounds__(256)
conv_stats_kernel(const float* __restrict__ pre, const float* __restrict__ rel,
                  const int* __restrict__ src_ids, const int* __restrict__ ridx,
                  const float* __restrict__ W, const float* __restrict__ bias,
                  float* __restrict__ gSum, float* __restrict__ gSq) {
  __shared__ float Wl[CIN * 3 * CCH];
  __shared__ float bl[CCH];
  __shared__ float sSum[CCH];
  __shared__ float sSq[CCH];
  int tid = threadIdx.x;
  for (int i = tid; i < CIN * 3 * CCH; i += 256) Wl[i] = W[i];
  if (tid < CCH) { bl[tid] = bias[tid]; sSum[tid] = 0.f; sSq[tid] = 0.f; }
  __syncthreads();

  int e = blockIdx.x;
  int d = tid;
  bool act = (d < DDIM);

  float xv[CIN][3];
  #pragma unroll
  for (int ci = 0; ci < CIN; ++ci)
    #pragma unroll
    for (int k = 0; k < 3; ++k) xv[ci][k] = 0.f;

  if (act) {
    const float* rows[CIN];
    rows[0] = pre + (size_t)src_ids[e] * DDIM;
    #pragma unroll
    for (int j = 1; j < CIN; ++j)
      rows[j] = rel + (size_t)ridx[(size_t)e * (CIN - 1) + (j - 1)] * DDIM;
    #pragma unroll
    for (int ci = 0; ci < CIN; ++ci)
      #pragma unroll
      for (int k = 0; k < 3; ++k) {
        int dd = d + k - 1;
        xv[ci][k] = (dd >= 0 && dd < DDIM) ? rows[ci][dd] : 0.f;
      }
  }

  int lane = tid & 31;
  for (int c = 0; c < CCH; ++c) {
    float y = bl[c];
    #pragma unroll
    for (int ci = 0; ci < CIN; ++ci)
      #pragma unroll
      for (int k = 0; k < 3; ++k)
        y = fmaf(Wl[c * CIN * 3 + ci * 3 + k], xv[ci][k], y);
    float v  = act ? y     : 0.f;
    float v2 = act ? y * y : 0.f;
    v  = wave_sum(v);
    v2 = wave_sum(v2);
    if (lane == 0) { atomicAdd(&sSum[c], v); atomicAdd(&sSq[c], v2); }
  }
  __syncthreads();
  if (tid < CCH) { atomicAdd(&gSum[tid], sSum[tid]); atomicAdd(&gSq[tid], sSq[tid]); }
}

// ---------------- conv pass 2: conv + BN1 + ReLU -> A_f16 [E][KP] ----------------
template<int CIN>
__global__ void __launch_bounds__(256)
conv_act_kernel(const float* __restrict__ pre, const float* __restrict__ rel,
                const int* __restrict__ src_ids, const int* __restrict__ ridx,
                const float* __restrict__ W, const float* __restrict__ bias,
                const float* __restrict__ gSum, const float* __restrict__ gSq,
                const float* __restrict__ g1, const float* __restrict__ be1,
                _Float16* __restrict__ A16) {
  __shared__ float Wl[CIN * 3 * CCH];
  __shared__ float bl[CCH];
  __shared__ float sA[CCH];
  __shared__ float sB[CCH];
  int tid = threadIdx.x;
  for (int i = tid; i < CIN * 3 * CCH; i += 256) Wl[i] = W[i];
  if (tid < CCH) {
    bl[tid] = bias[tid];
    const float inv = 1.0f / ((float)E_EDGES * (float)DDIM);
    float mu  = gSum[tid] * inv;
    float var = gSq[tid] * inv - mu * mu;
    float a   = g1[tid] * rsqrtf(var + EPS_BN);
    sA[tid] = a;
    sB[tid] = be1[tid] - mu * a;
  }
  __syncthreads();

  int e = blockIdx.x;
  int d = tid;
  if (d < DDIM) {
    float xv[CIN][3];
    const float* rows[CIN];
    rows[0] = pre + (size_t)src_ids[e] * DDIM;
    #pragma unroll
    for (int j = 1; j < CIN; ++j)
      rows[j] = rel + (size_t)ridx[(size_t)e * (CIN - 1) + (j - 1)] * DDIM;
    #pragma unroll
    for (int ci = 0; ci < CIN; ++ci)
      #pragma unroll
      for (int k = 0; k < 3; ++k) {
        int dd = d + k - 1;
        xv[ci][k] = (dd >= 0 && dd < DDIM) ? rows[ci][dd] : 0.f;
      }
    _Float16* arow = A16 + (size_t)e * KP;
    for (int c = 0; c < CCH; ++c) {
      float y = bl[c];
      #pragma unroll
      for (int ci = 0; ci < CIN; ++ci)
        #pragma unroll
        for (int k = 0; k < 3; ++k)
          y = fmaf(Wl[c * CIN * 3 + ci * 3 + k], xv[ci][k], y);
      float v = fmaxf(fmaf(sA[c], y, sB[c]), 0.f);
      arow[c * DDIM + d] = (_Float16)v;
    }
  } else if (d < DDIM + (KP - KDIM)) {
    A16[(size_t)e * KP + KDIM + (d - DDIM)] = (_Float16)0.f;
  }
}

// ---------------- WMMA GEMM: out[E][NP] = A_f16[E][KP] * B_f16[NP][KP]^T + fc_b ----------------
// Each wave computes a 32(M) x 16(N) output tile; 64*13 = 832 wave jobs, 4 waves/block.
__global__ void __launch_bounds__(128)
gemm_wmma_kernel(const _Float16* __restrict__ A, const _Float16* __restrict__ B,
                 const float* __restrict__ fcb, float* __restrict__ out) {
  int lane = threadIdx.x & 31;
  int wave = threadIdx.x >> 5;
  int job  = blockIdx.x * 4 + wave;        // 0..831
  int mt2  = job / 13;                     // 0..63 -> rows mt2*32 .. +31
  int nt   = job - mt2 * 13;               // 0..12 -> cols nt*16 .. +15

  int m   = lane & 15;
  int sel = lane >> 4;

  const _Float16* a0p = A + (size_t)(mt2 * 32 + m) * KP + 8 * sel;
  const _Float16* a1p = a0p + (size_t)16 * KP;
  const _Float16* bp  = B + (size_t)(nt * 16 + m) * KP + 8 * sel;

  v8f c0 = {};
  v8f c1 = {};
  for (int kb = 0; kb < KP / 32; ++kb) {
    int ko = kb * 32;
    v8h a0l = *(const v8h*)(a0p + ko);
    v8h a0h = *(const v8h*)(a0p + ko + 16);
    v8h a1l = *(const v8h*)(a1p + ko);
    v8h a1h = *(const v8h*)(a1p + ko + 16);
    v8h bl  = *(const v8h*)(bp + ko);
    v8h bh  = *(const v8h*)(bp + ko + 16);
    v16h av0 = cat16(a0l, a0h);
    v16h av1 = cat16(a1l, a1h);
    v16h bv  = cat16(bl, bh);
    c0 = __builtin_amdgcn_wmma_f32_16x16x32_f16(false, av0, false, bv,
                                                (short)0, c0, false, false);
    c1 = __builtin_amdgcn_wmma_f32_16x16x32_f16(false, av1, false, bv,
                                                (short)0, c1, false, false);
  }

  int n = nt * 16 + m;
  float bias = (n < DDIM) ? fcb[n] : 0.f;
  #pragma unroll
  for (int r = 0; r < 8; ++r) {
    int mm = mt2 * 32 + r + 8 * sel;
    out[(size_t)mm * NP + n]        = c0[r] + bias;
    out[(size_t)(mm + 16) * NP + n] = c1[r] + bias;
  }
}

// ---------------- BN2 stats: one block per feature n ----------------
__global__ void __launch_bounds__(256)
bn2_stats_kernel(const float* __restrict__ gout, const float* __restrict__ g2,
                 const float* __restrict__ be2, float* __restrict__ alpha2,
                 float* __restrict__ beta2) {
  __shared__ float ls[256];
  __shared__ float ls2[256];
  int n = blockIdx.x;                      // 0..199
  int tid = threadIdx.x;
  float s = 0.f, s2 = 0.f;
  for (int e = tid; e < E_EDGES; e += 256) {
    float v = gout[(size_t)e * NP + n];
    s += v; s2 += v * v;
  }
  ls[tid] = s; ls2[tid] = s2;
  __syncthreads();
  #pragma unroll
  for (int off = 128; off > 0; off >>= 1) {
    if (tid < off) { ls[tid] += ls[tid + off]; ls2[tid] += ls2[tid + off]; }
    __syncthreads();
  }
  if (tid == 0) {
    float mean = ls[0] / (float)E_EDGES;
    float var  = ls2[0] / (float)E_EDGES - mean * mean;
    float a = g2[n] * rsqrtf(var + EPS_BN);
    alpha2[n] = a;
    beta2[n]  = be2[n] - mean * a;
  }
}

// ---------------- BN2 apply + ReLU -> xact [E][D] ----------------
__global__ void bn2_act_kernel(const float* __restrict__ gout,
                               const float* __restrict__ alpha2,
                               const float* __restrict__ beta2,
                               float* __restrict__ xa) {
  int i = blockIdx.x * blockDim.x + threadIdx.x;
  if (i >= E_EDGES * DDIM) return;
  int e = i / DDIM;
  int n = i - e * DDIM;
  float v = fmaf(alpha2[n], gout[(size_t)e * NP + n], beta2[n]);
  xa[i] = fmaxf(v, 0.f);
}

// ---------------- cosine: one wave per edge ----------------
__global__ void __launch_bounds__(256)
cos_kernel(const float* __restrict__ x1, const float* __restrict__ x2,
           const float* __restrict__ x3, const float* __restrict__ re,
           float* __restrict__ cbuf) {
  int lane = threadIdx.x & 31;
  int wave = threadIdx.x >> 5;
  int e = blockIdx.x * 8 + wave;           // 256 blocks * 8 waves = 2048
  const float* a1 = x1 + (size_t)e * DDIM;
  const float* a2 = x2 + (size_t)e * DDIM;
  const float* a3 = x3 + (size_t)e * DDIM;
  const float* ar = re + (size_t)e * DDIM;
  float d1 = 0.f, d2 = 0.f, d3 = 0.f, n1 = 0.f, n2 = 0.f, n3 = 0.f, nr = 0.f;
  for (int n = lane; n < DDIM; n += 32) {
    float r = ar[n];
    float v1 = a1[n], v2 = a2[n], v3 = a3[n];
    d1 = fmaf(v1, r, d1); d2 = fmaf(v2, r, d2); d3 = fmaf(v3, r, d3);
    n1 = fmaf(v1, v1, n1); n2 = fmaf(v2, v2, n2); n3 = fmaf(v3, v3, n3);
    nr = fmaf(r, r, nr);
  }
  d1 = wave_sum(d1); d2 = wave_sum(d2); d3 = wave_sum(d3);
  n1 = wave_sum(n1); n2 = wave_sum(n2); n3 = wave_sum(n3);
  nr = wave_sum(nr);
  if (lane == 0) {
    float sr = sqrtf(nr);
    float c = d1 / fmaxf(sqrtf(n1) * sr, EPS_COS)
            + d2 / fmaxf(sqrtf(n2) * sr, EPS_COS)
            + d3 / fmaxf(sqrtf(n3) * sr, EPS_COS);
    cbuf[e] = c;
  }
}

// ---------------- scatter ----------------
__global__ void scatter_kernel(const float* __restrict__ cbuf,
                               const int* __restrict__ o,
                               float* __restrict__ out) {
  int i = blockIdx.x * blockDim.x + threadIdx.x;
  if (i < E_EDGES) out[(size_t)i * NNODES + o[i]] = cbuf[i];
}

// ---------------- host side ----------------
// workspace layout (bytes, 256-aligned)
#define A16_OFF   ((size_t)0)
#define A16_BYTES ((size_t)E_EDGES * KP * 2)                 // 41,025,536
#define B16_OFF   (A16_OFF + A16_BYTES)
#define B16_BYTES ((size_t)NP * KP * 2)                      //  4,166,656
#define GOUT_OFF  (B16_OFF + B16_BYTES)
#define GOUT_BYTES ((size_t)E_EDGES * NP * 4)                //  1,703,936
#define XACT_OFF  (GOUT_OFF + GOUT_BYTES)
#define XACT_ONE  ((size_t)E_EDGES * DDIM * 4)               //  1,638,400
#define STATS1_OFF (XACT_OFF + 4 * XACT_ONE)
#define STATS1_BYTES ((size_t)512)                           // 2*50 floats
#define STATS2_OFF (STATS1_OFF + STATS1_BYTES)
#define STATS2_BYTES ((size_t)2048)                          // 2*200 floats
#define CBUF_OFF  (STATS2_OFF + STATS2_BYTES)
#define CBUF_BYTES ((size_t)E_EDGES * 4)
#define WS_NEEDED (CBUF_OFF + CBUF_BYTES)

struct BlockCfg {
  int cin;
  const int* ridx;
  const float* W;
  const float* b;
};

static void run_block(const BlockCfg& cfg,
                      const float* pre, const float* rel, const int* src,
                      const float* g1, const float* be1,
                      const float* g2, const float* be2,
                      const float* fcb,
                      _Float16* A16, const _Float16* B16,
                      float* gout, float* stats1, float* stats2,
                      float* xact, hipStream_t stream) {
  float* gSum = stats1;
  float* gSq  = stats1 + CCH;
  float* alpha2 = stats2;
  float* beta2  = stats2 + DDIM;

  zero_f32<<<1, 128, 0, stream>>>(stats1, 2 * CCH);

  switch (cfg.cin) {
    case 2:
      conv_stats_kernel<2><<<E_EDGES, 256, 0, stream>>>(pre, rel, src, cfg.ridx,
          cfg.W, cfg.b, gSum, gSq);
      conv_act_kernel<2><<<E_EDGES, 256, 0, stream>>>(pre, rel, src, cfg.ridx,
          cfg.W, cfg.b, gSum, gSq, g1, be1, A16);
      break;
    case 3:
      conv_stats_kernel<3><<<E_EDGES, 256, 0, stream>>>(pre, rel, src, cfg.ridx,
          cfg.W, cfg.b, gSum, gSq);
      conv_act_kernel<3><<<E_EDGES, 256, 0, stream>>>(pre, rel, src, cfg.ridx,
          cfg.W, cfg.b, gSum, gSq, g1, be1, A16);
      break;
    default:
      conv_stats_kernel<4><<<E_EDGES, 256, 0, stream>>>(pre, rel, src, cfg.ridx,
          cfg.W, cfg.b, gSum, gSq);
      conv_act_kernel<4><<<E_EDGES, 256, 0, stream>>>(pre, rel, src, cfg.ridx,
          cfg.W, cfg.b, gSum, gSq, g1, be1, A16);
      break;
  }

  gemm_wmma_kernel<<<208, 128, 0, stream>>>(A16, B16, fcb, gout);
  bn2_stats_kernel<<<DDIM, 256, 0, stream>>>(gout, g2, be2, alpha2, beta2);
  bn2_act_kernel<<<(E_EDGES * DDIM + 255) / 256, 256, 0, stream>>>(gout, alpha2,
                                                                   beta2, xact);
}

extern "C" void kernel_launch(void* const* d_in, const int* in_sizes, int n_in,
                              void* d_out, int out_size, void* d_ws, size_t ws_size,
                              hipStream_t stream) {
  const float* pre  = (const float*)d_in[0];
  const float* rel  = (const float*)d_in[1];
  const float* W2   = (const float*)d_in[2];
  const float* b2   = (const float*)d_in[3];
  const float* W3   = (const float*)d_in[4];
  const float* b3   = (const float*)d_in[5];
  const float* W4   = (const float*)d_in[6];
  const float* b4   = (const float*)d_in[7];
  const float* fcw  = (const float*)d_in[8];
  const float* fcb  = (const float*)d_in[9];
  const float* g1   = (const float*)d_in[10];
  const float* be1  = (const float*)d_in[11];
  const float* g2   = (const float*)d_in[12];
  const float* be2  = (const float*)d_in[13];
  const int* src    = (const int*)d_in[14];
  const int* etype  = (const int*)d_in[15];
  const int* p1     = (const int*)d_in[16];
  const int* p2     = (const int*)d_in[17];
  const int* p3     = (const int*)d_in[18];
  const int* o      = (const int*)d_in[19];

  if (ws_size < WS_NEEDED) return;

  char* wsb = (char*)d_ws;
  _Float16* A16 = (_Float16*)(wsb + A16_OFF);
  _Float16* B16 = (_Float16*)(wsb + B16_OFF);
  float* gout   = (float*)(wsb + GOUT_OFF);
  float* xact   = (float*)(wsb + XACT_OFF);   // 4 consecutive [E][D] buffers
  float* stats1 = (float*)(wsb + STATS1_OFF);
  float* stats2 = (float*)(wsb + STATS2_OFF);
  float* cbuf   = (float*)(wsb + CBUF_OFF);
  float* out    = (float*)d_out;

  // fc_w -> f16, padded
  {
    long total = (long)NP * KP;
    int blocks = (int)((total + 255) / 256);
    convert_b_kernel<<<blocks, 256, 0, stream>>>(fcw, B16);
  }

  const size_t xone = (size_t)E_EDGES * DDIM;
  BlockCfg cfgs[4] = {
    {2, p1,    W2, b2},   // x1
    {3, p2,    W3, b3},   // x2
    {4, p3,    W4, b4},   // x3
    {2, etype, W2, b2},   // r_e
  };
  for (int bi = 0; bi < 4; ++bi) {
    run_block(cfgs[bi], pre, rel, src, g1, be1, g2, be2, fcb,
              A16, B16, gout, stats1, stats2, xact + (size_t)bi * xone, stream);
  }

  // cosine sums
  cos_kernel<<<E_EDGES / 8, 256, 0, stream>>>(xact, xact + xone, xact + 2 * xone,
                                              xact + 3 * xone, cbuf);

  // zero output then scatter
  int n4 = out_size >> 2;   // out_size = 2048*40943, divisible by 4
  zero_f32x4<<<(n4 + 255) / 256, 256, 0, stream>>>((float4*)out, n4);
  scatter_kernel<<<(E_EDGES + 255) / 256, 256, 0, stream>>>(cbuf, o, out);
}
